// NeuroSAT_18940805776105
// MI455X (gfx1250) — matrix operand order repaired and checked by reading, not compiled, not run on hardware
//
#include <hip/hip_runtime.h>
#include <hip/hip_bf16.h>

typedef __attribute__((ext_vector_type(16))) _Float16 v16h;
typedef __attribute__((ext_vector_type(8)))  _Float16 v8h;
typedef __attribute__((ext_vector_type(8)))  float    v8f;

#define HDIM 64

__device__ __forceinline__ float sigm(float v) { return 1.0f / (1.0f + __expf(-v)); }

// ---------------------------------------------------------------------------
// Pack a KxM row-major f32 weight matrix into WMMA B-fragment order (f16).
// frag = kt*(M/16)+tile; lane holds column tile*16+lane%16,
// halves j=0..15 map to Krel = (lane>=16 ? 8 : 0) + (j<8 ? j : 8 + j) within the 32-deep tile.
// ---------------------------------------------------------------------------
__global__ void pack_w_kernel(const float* __restrict__ Wsrc, _Float16* __restrict__ P,
                              int K, int M) {
  int t = blockIdx.x * blockDim.x + threadIdx.x;
  int Mt = M >> 4;
  int frags = (K >> 5) * Mt;
  if (t >= frags * 32) return;
  int frag = t >> 5, lane = t & 31;
  int kt = frag / Mt, tile = frag % Mt;
  int col = tile * 16 + (lane & 15);
  int kbase = kt * 32 + ((lane >> 4) << 3);
  _Float16* dst = P + (size_t)t * 16;
#pragma unroll
  for (int j = 0; j < 16; ++j)
    dst[j] = (_Float16)Wsrc[(size_t)(kbase + (j < 8 ? j : 8 + j)) * M + col];
}

// ---------------------------------------------------------------------------
// Compose two 64x64 linear layers:  C = A@B ;  Cb = Ab@B + Bb
// ---------------------------------------------------------------------------
__global__ void compose64_kernel(const float* __restrict__ A, const float* __restrict__ Ab,
                                 const float* __restrict__ B, const float* __restrict__ Bb,
                                 float* __restrict__ C, float* __restrict__ Cb) {
  int idx = blockIdx.x * blockDim.x + threadIdx.x;
  if (idx >= 64 * 64) return;
  int i = idx >> 6, j = idx & 63;
  float s = 0.f;
  for (int k = 0; k < 64; ++k) s += A[i * 64 + k] * B[k * 64 + j];
  C[idx] = s;
  if (i == 0) {
    float t2 = 0.f;
    for (int k = 0; k < 64; ++k) t2 += Ab[k] * B[k * 64 + j];
    Cb[j] = t2 + Bb[j];
  }
}

// ---------------------------------------------------------------------------
// WMMA GEMM:  Y[N x M] = sum_i Ai[N x 64] @ Wi(packed) + bias0 (+ bias1)
// Block = 8 waves, all sharing one column group (blockIdx.y); each wave owns a
// 16-row x 64-col strip.  Packed weights for this column group (up to 3 x 8 KB)
// are staged in LDS once per block and served to WMMA from ds_load_b128.
// A0 may be row-permuted via perm0 (gather fold for out[flip_perm]).
// ---------------------------------------------------------------------------
__global__ void gemm_wmma_k64(
    const _Float16* __restrict__ A0, const int* __restrict__ perm0,
    const _Float16* __restrict__ W0,
    const _Float16* __restrict__ A1, const _Float16* __restrict__ W1,
    const _Float16* __restrict__ A2, const _Float16* __restrict__ W2,
    const float* __restrict__ bias0, const float* __restrict__ bias1,
    float* __restrict__ Y, int M, int rowTiles) {
  __shared__ __align__(16) _Float16 lds[3 * 4096];  // 3 inputs x 8 frags x 512 halves

  int tid = threadIdx.x;
  int lane = tid & 31;
  int waveId = tid >> 5;
  int cg = blockIdx.y;
  int Mtiles = M >> 4;

  const _Float16* As[3] = {A0, A1, A2};
  const _Float16* Ws[3] = {W0, W1, W2};

  // ---- cooperative stage of this column-group's weight fragments into LDS ----
#pragma unroll
  for (int inp = 0; inp < 3; ++inp) {
    if (!Ws[inp]) continue;
    // 8 fragments (kt 0..1 x ct 0..3) of 512 halves; copy as 512 x 16B chunks
    for (int c = tid; c < 512; c += 256) {
      int f = c >> 6;             // fragment 0..7 (= kt*4+ct)
      int kt = f >> 2, ct = f & 3;
      int off = (c & 63) * 8;     // halves within fragment
      const _Float16* s = Ws[inp] + (size_t)(kt * Mtiles + cg * 4 + ct) * 512 + off;
      *(v8h*)(&lds[inp * 4096 + f * 512 + off]) = *(const v8h*)s;
    }
  }
  __syncthreads();

  int rt = blockIdx.x * 8 + waveId;
  if (rt >= rowTiles) return;

  int halfSel = lane >> 4;   // 0: lanes 0-15, 1: lanes 16-31
  int rit = lane & 15;
  int r = rt * 16 + rit;

  v8f acc[4] = {};
#pragma unroll
  for (int inp = 0; inp < 3; ++inp) {
    const _Float16* A = As[inp];
    if (!A) continue;
    int rowA = (inp == 0 && perm0) ? perm0[r] : r;
    const _Float16* abase = A + (size_t)rowA * HDIM + (halfSel << 3);
#pragma unroll
    for (int kt = 0; kt < 2; ++kt) {
      // ISA 16-bit A layout: halves [0..7] -> K = kt*32 + halfSel*8 + i
      //                      halves [8..15]-> K = kt*32 + halfSel*8 + 16 + i
      v8h lo = *(const v8h*)(abase + kt * 32);
      v8h hi = *(const v8h*)(abase + kt * 32 + 16);
      v16h a;
#pragma unroll
      for (int i2 = 0; i2 < 8; ++i2) { a[i2] = lo[i2]; a[8 + i2] = hi[i2]; }
#pragma unroll
      for (int ct = 0; ct < 4; ++ct) {
        v16h b = *(const v16h*)(&lds[inp * 4096 + (kt * 4 + ct) * 512 + lane * 16]);
        acc[ct] = __builtin_amdgcn_wmma_f32_16x16x32_f16(
            false, a, false, b, (short)0, acc[ct], false, false);
      }
    }
  }
  // C/D layout: VGPR v -> row rt*16 + halfSel*8 + v, col = tile*16 + lane%16
  int rbase = rt * 16 + (halfSel << 3);
#pragma unroll
  for (int ct = 0; ct < 4; ++ct) {
    int col = (cg * 4 + ct) * 16 + rit;
    float bv = bias0 ? bias0[col] : 0.f;
    if (bias1) bv += bias1[col];
#pragma unroll
    for (int v = 0; v < 8; ++v)
      Y[(size_t)(rbase + v) * M + col] = acc[ct][v] + bv;
  }
}

// ---------------------------------------------------------------------------
// segment_sum pieces
// ---------------------------------------------------------------------------
__global__ void prop_self_kernel(const float* __restrict__ src, float* __restrict__ dst, int n) {
  int t = blockIdx.x * blockDim.x + threadIdx.x;
  if (t < n) dst[t] = src[t];  // self loops
}

// first-half edges are grouped: each clause dst receives exactly 12 literal srcs
__global__ void prop_clause_kernel(const float* __restrict__ mlp, float* __restrict__ msg,
                                   const int* __restrict__ esrc, const int* __restrict__ edst,
                                   int ncls) {
  int t = blockIdx.x * blockDim.x + threadIdx.x;
  if (t >= ncls * HDIM) return;
  int ci = t >> 6, j = t & 63;
  int base = ci * 12;
  int d = edst[base];                       // constant across the 12 edges
  float s = mlp[(size_t)d * HDIM + j];      // self loop
#pragma unroll
  for (int k = 0; k < 12; ++k) {
    int s0 = esrc[base + k];
    s += mlp[(size_t)s0 * HDIM + j];
  }
  msg[(size_t)d * HDIM + j] = s;
}

// second-half edges: clause -> literal, random dst => atomics
__global__ void prop_lit_kernel(const float* __restrict__ mlp, float* __restrict__ msg,
                                const int* __restrict__ esrc, const int* __restrict__ edst,
                                int ne) {
  int t = blockIdx.x * blockDim.x + threadIdx.x;
  int e = t >> 6, j = t & 63;
  if (e >= ne) return;
  int s0 = esrc[e], d0 = edst[e];
  atomicAdd(&msg[(size_t)d0 * HDIM + j], mlp[(size_t)s0 * HDIM + j]);
}

__global__ void cvt_f16_kernel(const float* __restrict__ src, _Float16* __restrict__ dst, int n) {
  int t = blockIdx.x * blockDim.x + threadIdx.x;
  if (t < n) dst[t] = (_Float16)src[t];
}

// ---------------------------------------------------------------------------
// LSTM elementwise updates (gates layout: [i | f | g | o] of 64 each)
// ---------------------------------------------------------------------------
__global__ void lstm_c_kernel(const float* __restrict__ g, const float* __restrict__ out,
                              const float* __restrict__ mk, float* __restrict__ ch,
                              float* __restrict__ cc, _Float16* __restrict__ ch16,
                              _Float16* __restrict__ tmp16, int N) {
  int t = blockIdx.x * blockDim.x + threadIdx.x;
  if (t >= N * HDIM) return;
  int n = t >> 6, j = t & 63;
  const float* gr = g + (size_t)n * 256;
  float gi = gr[j], gf = gr[64 + j], gg = gr[128 + j], go = gr[192 + j];
  float c = sigm(gf) * cc[t] + sigm(gi) * tanhf(gg);
  float h = sigm(go) * tanhf(c);
  cc[t] = c; ch[t] = h; ch16[t] = (_Float16)h;
  float m = mk[n];
  tmp16[t] = (_Float16)((1.f - m) * out[t] + m * h);   // temp blend for c-MLP
}

__global__ void lstm_l_kernel(const float* __restrict__ g, const float* __restrict__ ch,
                              const float* __restrict__ mk, float* __restrict__ lh,
                              float* __restrict__ lc, _Float16* __restrict__ lh16,
                              float* __restrict__ out, _Float16* __restrict__ out16, int N) {
  int t = blockIdx.x * blockDim.x + threadIdx.x;
  if (t >= N * HDIM) return;
  int n = t >> 6, j = t & 63;
  const float* gr = g + (size_t)n * 256;
  float gi = gr[j], gf = gr[64 + j], gg = gr[128 + j], go = gr[192 + j];
  float c = sigm(gf) * lc[t] + sigm(gi) * tanhf(gg);
  float h = sigm(go) * tanhf(c);
  lc[t] = c; lh[t] = h; lh16[t] = (_Float16)h;
  float m = mk[n];
  float o = (1.f - m) * h + m * ch[t];
  out[t] = o; out16[t] = (_Float16)o;
}

// ---------------------------------------------------------------------------
// init: out0 = (x@li_w+li_b)*(1-m) + (x@ci_w+ci_b)*m ; zero LSTM state
// ---------------------------------------------------------------------------
__global__ void init_kernel(const float* __restrict__ x, const float* __restrict__ mk,
                            const float* __restrict__ liw, const float* __restrict__ lib,
                            const float* __restrict__ ciw, const float* __restrict__ cib,
                            float* __restrict__ out, _Float16* __restrict__ out16,
                            float* __restrict__ ch, float* __restrict__ cc,
                            float* __restrict__ lh, float* __restrict__ lc,
                            _Float16* __restrict__ ch16, _Float16* __restrict__ lh16, int N) {
  int t = blockIdx.x * blockDim.x + threadIdx.x;
  if (t >= N * HDIM) return;
  int n = t >> 6, j = t & 63;
  float x0 = x[2 * n], x1 = x[2 * n + 1];
  float lo = x0 * liw[j] + x1 * liw[64 + j] + lib[j];
  float co = x0 * ciw[j] + x1 * ciw[64 + j] + cib[j];
  float m = mk[n];
  float o = lo * (1.f - m) + co * m;
  out[t] = o; out16[t] = (_Float16)o;
  ch[t] = 0.f; cc[t] = 0.f; lh[t] = 0.f; lc[t] = 0.f;
  ch16[t] = (_Float16)0.f; lh16[t] = (_Float16)0.f;
}

__global__ void vote_pre_kernel(const float* __restrict__ out, const float* __restrict__ mk,
                                _Float16* __restrict__ vin16, int N) {
  int t = blockIdx.x * blockDim.x + threadIdx.x;
  if (t >= N * HDIM) return;
  int n = t >> 6;
  vin16[t] = (_Float16)(out[t] * (1.f - mk[n]));
}

__global__ void vote_final_kernel(const float* __restrict__ vbuf, const float* __restrict__ w3,
                                  const float* __restrict__ b3, const float* __restrict__ mk,
                                  float* __restrict__ dout, int N) {
  int n = blockIdx.x * blockDim.x + threadIdx.x;
  if (n >= N) return;
  float s = b3[0];
  for (int j = 0; j < HDIM; ++j) s += vbuf[(size_t)n * HDIM + j] * w3[j];
  dout[n] = s * (1.f - mk[n]);
}

// ---------------------------------------------------------------------------
extern "C" void kernel_launch(void* const* d_in, const int* in_sizes, int n_in,
                              void* d_out, int out_size, void* d_ws, size_t ws_size,
                              hipStream_t stream) {
  const float* x      = (const float*)d_in[0];
  const float* mask   = (const float*)d_in[1];
  const int*   ei     = (const int*)d_in[2];
  const int*   flip   = (const int*)d_in[3];
  const float* li_w   = (const float*)d_in[4];
  const float* li_b   = (const float*)d_in[5];
  const float* ci_w   = (const float*)d_in[6];
  const float* ci_b   = (const float*)d_in[7];
  const float* lm1_w  = (const float*)d_in[8];
  const float* lm1_b  = (const float*)d_in[9];
  const float* lm2_w  = (const float*)d_in[10];
  const float* lm2_b  = (const float*)d_in[11];
  const float* lm3_w  = (const float*)d_in[12];
  const float* lm3_b  = (const float*)d_in[13];
  const float* cm1_w  = (const float*)d_in[14];
  const float* cm1_b  = (const float*)d_in[15];
  const float* cm2_w  = (const float*)d_in[16];
  const float* cm2_b  = (const float*)d_in[17];
  const float* cm3_w  = (const float*)d_in[18];
  const float* cm3_b  = (const float*)d_in[19];
  const float* lu_wih = (const float*)d_in[20];
  const float* lu_whh = (const float*)d_in[21];
  const float* lu_bih = (const float*)d_in[22];
  const float* lu_bhh = (const float*)d_in[23];
  const float* cu_wih = (const float*)d_in[24];
  const float* cu_whh = (const float*)d_in[25];
  const float* cu_bih = (const float*)d_in[26];
  const float* cu_bhh = (const float*)d_in[27];
  const float* lv1_w  = (const float*)d_in[28];
  const float* lv1_b  = (const float*)d_in[29];
  const float* lv2_w  = (const float*)d_in[30];
  const float* lv2_b  = (const float*)d_in[31];
  const float* lv3_w  = (const float*)d_in[32];
  const float* lv3_b  = (const float*)d_in[33];

  const int N  = in_sizes[1];        // 19840 (divisible by 16)
  const int TE = in_sizes[2] / 2;    // total edges (both halves)
  const int E1 = TE / 2;             // lit->cls half
  const int E2 = TE - E1;            // cls->lit half
  const int NCLS = E1 / 12;
  const int rowTiles = N / 16;
  const size_t F = (size_t)N * HDIM;

  // ---- workspace carve-up (f32 region then f16 region) ----
  float* W       = (float*)d_ws;
  float* out_f   = W;
  float* ch      = W + 1 * F;
  float* cc      = W + 2 * F;
  float* lh      = W + 3 * F;
  float* lc      = W + 4 * F;
  float* mlpout  = W + 5 * F;
  float* msg     = W + 6 * F;
  float* gbuf    = W + 7 * F;          // N x 256
  float* smallf  = W + 11 * F;
  float* T1  = smallf;                  // 64x64 compose scratch
  float* Tb1 = smallf + 4096;           // 64
  float* Cc  = smallf + 4160;           // 64x64 composed weights scratch
  float* LMb = smallf + 8256;           // persistent composed biases
  float* CMb = smallf + 8320;
  float* LVb = smallf + 8384;
  _Float16* Hb     = (_Float16*)(smallf + 8448);
  _Float16* out16  = Hb;
  _Float16* ch16   = Hb + 1 * F;
  _Float16* lh16   = Hb + 2 * F;
  _Float16* msg16  = Hb + 3 * F;
  _Float16* tmp16  = Hb + 4 * F;       // temp blend / vote input (reused)
  _Float16* LMWp   = Hb + 5 * F;       // packed weights
  _Float16* CMWp   = LMWp + 4096;
  _Float16* LVWp   = CMWp + 4096;
  _Float16* CUWIHp = LVWp + 4096;
  _Float16* CUWHHp = CUWIHp + 16384;
  _Float16* LUWIHAp= CUWHHp + 16384;
  _Float16* LUWIHBp= LUWIHAp + 16384;
  _Float16* LUWHHp = LUWIHBp + 16384;

  const int TPB = 256;
  const int gElem = (int)((F + TPB - 1) / TPB);

  // ---- setup: compose linear MLPs, pack all GEMM weights to f16 fragments ----
  compose64_kernel<<<16, TPB, 0, stream>>>(lm1_w, lm1_b, lm2_w, lm2_b, T1, Tb1);
  compose64_kernel<<<16, TPB, 0, stream>>>(T1, Tb1, lm3_w, lm3_b, Cc, LMb);
  pack_w_kernel<<<1, TPB, 0, stream>>>(Cc, LMWp, 64, 64);
  compose64_kernel<<<16, TPB, 0, stream>>>(cm1_w, cm1_b, cm2_w, cm2_b, T1, Tb1);
  compose64_kernel<<<16, TPB, 0, stream>>>(T1, Tb1, cm3_w, cm3_b, Cc, CMb);
  pack_w_kernel<<<1, TPB, 0, stream>>>(Cc, CMWp, 64, 64);
  compose64_kernel<<<16, TPB, 0, stream>>>(lv1_w, lv1_b, lv2_w, lv2_b, Cc, LVb);
  pack_w_kernel<<<1, TPB, 0, stream>>>(Cc, LVWp, 64, 64);
  pack_w_kernel<<<4, TPB, 0, stream>>>(cu_wih, CUWIHp, 64, 256);
  pack_w_kernel<<<4, TPB, 0, stream>>>(cu_whh, CUWHHp, 64, 256);
  pack_w_kernel<<<4, TPB, 0, stream>>>(lu_wih, LUWIHAp, 64, 256);            // rows 0..63
  pack_w_kernel<<<4, TPB, 0, stream>>>(lu_wih + 64 * 256, LUWIHBp, 64, 256); // rows 64..127
  pack_w_kernel<<<4, TPB, 0, stream>>>(lu_whh, LUWHHp, 64, 256);

  init_kernel<<<gElem, TPB, 0, stream>>>(x, mask, li_w, li_b, ci_w, ci_b,
                                         out_f, out16, ch, cc, lh, lc, ch16, lh16, N);

  auto launch_gemm = [&](const _Float16* A0, const int* perm, const _Float16* W0,
                         const _Float16* A1, const _Float16* W1,
                         const _Float16* A2, const _Float16* W2,
                         const float* b0, const float* b1, float* Yp, int M) {
    int cgc = M / 64;
    dim3 grid((rowTiles + 7) / 8, cgc);   // 8 waves/block, all on one column group
    gemm_wmma_k64<<<grid, TPB, 0, stream>>>(A0, perm, W0, A1, W1, A2, W2,
                                            b0, b1, Yp, M, rowTiles);
  };
  auto launch_prop = [&]() {
    prop_self_kernel<<<gElem, TPB, 0, stream>>>(mlpout, msg, (int)F);
    prop_clause_kernel<<<(NCLS * HDIM + TPB - 1) / TPB, TPB, 0, stream>>>(
        mlpout, msg, ei, ei + TE, NCLS);
    prop_lit_kernel<<<(E2 * HDIM + TPB - 1) / TPB, TPB, 0, stream>>>(
        mlpout, msg, ei + E1, ei + TE + E1, E2);
    cvt_f16_kernel<<<gElem, TPB, 0, stream>>>(msg, msg16, (int)F);
  };

  for (int it = 0; it < 23; ++it) {  // ITERS - 1
    // l_msg = prop(composed l-MLP(out))
    launch_gemm(out16, nullptr, LMWp, nullptr, nullptr, nullptr, nullptr,
                LMb, nullptr, mlpout, 64);
    launch_prop();
    // ch, cc = LSTM_c(l_msg, ch, cc);  tmp = blend(out, ch)
    launch_gemm(msg16, nullptr, CUWIHp, ch16, CUWHHp, nullptr, nullptr,
                cu_bih, cu_bhh, gbuf, 256);
    lstm_c_kernel<<<gElem, TPB, 0, stream>>>(gbuf, out_f, mask, ch, cc, ch16, tmp16, N);
    // c_msg = prop(composed c-MLP(tmp))
    launch_gemm(tmp16, nullptr, CMWp, nullptr, nullptr, nullptr, nullptr,
                CMb, nullptr, mlpout, 64);
    launch_prop();
    // lh, lc = LSTM_l([out[flip], c_msg], lh, lc);  out = blend(lh, ch)
    launch_gemm(out16, flip, LUWIHAp, msg16, LUWIHBp, lh16, LUWHHp,
                lu_bih, lu_bhh, gbuf, 256);
    lstm_l_kernel<<<gElem, TPB, 0, stream>>>(gbuf, ch, mask, lh, lc, lh16, out_f, out16, N);
  }

  // vote = ((out*(1-m)) @ LVc + LVb) @ lv3 + lv3_b, masked
  vote_pre_kernel<<<gElem, TPB, 0, stream>>>(out_f, mask, tmp16, N);
  launch_gemm(tmp16, nullptr, LVWp, nullptr, nullptr, nullptr, nullptr,
              LVb, nullptr, mlpout, 64);
  vote_final_kernel<<<(N + TPB - 1) / TPB, TPB, 0, stream>>>(
      mlpout, lv3_w, lv3_b, mask, (float*)d_out, N);
}
